// SequencePredictor_80315888435785
// MI455X (gfx1250) — compile-verified, matrix operand order checked by reading, not run on hardware
//
#include <hip/hip_runtime.h>
#include <hip/hip_bf16.h>

// ---------------------------------------------------------------------------
// Seq2seq (biLSTM encoder + Bahdanau attention decoder) for MI455X / gfx1250.
// All matmuls use v_wmma_f32_16x16x32_bf16 (bf16 inputs, f32 accumulate).
// Encoder: persistent kernel, 1 workgroup (512 thr / 16 waves) per direction,
//          x_t staged into LDS with CDNA5 async global->LDS DMA
//          (global_load_async_to_lds_b128 / s_wait_asynccnt), LSTM cell state
//          in registers across all 512 steps, redundancy-1 weight streaming
//          from L2 (laundered *offset* defeats LICM without destroying the
//          pointer's global address-space provenance -> global_load_b128).
// Decoder: 4 kernels per step (cell GEMM / pq GEMM / score+softmax+ctx /
//          attention GEMM), recurrence ordered by the stream.
// keys and logits GEMMs hoisted out of the recurrences and run GPU-wide.
// ---------------------------------------------------------------------------

typedef __bf16 bf16_t;
typedef __attribute__((ext_vector_type(16))) __bf16 v16bf;
typedef __attribute__((ext_vector_type(8)))  __bf16 v8bf;
typedef __attribute__((ext_vector_type(8)))  float  v8f;

#define B_   64
#define TX_  512
#define TY_  256
#define V_   10000
#define DE_  256
#define HE_  256
#define HD_  512
#define AU_  256
#define AT_  256
#define VPAD 10016   // V padded to multiple of 16
#define NTP  (VPAD/16)

__device__ __forceinline__ float sigmoidf_(float x) {
    return 1.0f / (1.0f + __expf(-x));
}

// Load one 16x32 bf16 A/B fragment (row-major tile, leading dim ld elements).
// CDNA5 16-bit A layout: lanes 0-15 hold M=lane, K chunks {0..7,16..23};
// lanes 16-31 hold M=lane-16, K chunks {8..15,24..31}. Two 16B loads/lane.
__device__ __forceinline__ v16bf load_frag(const bf16_t* p, int ld) {
    int lane = threadIdx.x & 31;
    int r    = lane & 15;
    int kb   = (lane & 16) ? 8 : 0;
    const bf16_t* q = p + (size_t)r * ld + kb;
    v8bf lo = *(const v8bf*)(q);
    v8bf hi = *(const v8bf*)(q + 16);
    return __builtin_shufflevector(lo, hi, 0,1,2,3,4,5,6,7,8,9,10,11,12,13,14,15);
}

__device__ __forceinline__ v8f wmma_bf(v16bf a, v16bf b, v8f c) {
    return __builtin_amdgcn_wmma_f32_16x16x32_bf16(false, a, false, b, (short)0, c, false, false);
}

// --------------------------- weight prepack --------------------------------
// f32 W[K][N] -> bf16 Wt[Npad][K] (N-major so B fragments load like A of W^T)
__global__ void prepack_w(const float* __restrict__ W, bf16_t* __restrict__ out,
                          int K, int N, int Npad) {
    size_t idx = (size_t)blockIdx.x * 256 + threadIdx.x;
    size_t total = (size_t)Npad * K;
    if (idx >= total) return;
    int n = (int)(idx / K), k = (int)(idx % K);
    float v = (n < N) ? W[(size_t)k * N + n] : 0.0f;
    out[idx] = (bf16_t)v;
}

// ----------------------------- embedding -----------------------------------
// ids[B][T], E[V][DE] f32 -> out[T][B][DE] bf16 (time-major for the scans)
__global__ void embed_g(const int* __restrict__ ids, const float* __restrict__ E,
                        bf16_t* __restrict__ out, int T) {
    size_t idx = (size_t)blockIdx.x * 256 + threadIdx.x;
    int d = (int)(idx & 255);
    size_t tb = idx >> 8;
    int b = (int)(tb & 63);
    int t = (int)(tb >> 6);
    int id = ids[(size_t)b * T + t];
    out[idx] = (bf16_t)E[(size_t)id * DE_ + d];
}

// ------------------------- persistent encoder ------------------------------
// grid = 2 (dir), block = 512 (16 waves, 4/SIMD -> full VGPR budget).
// Wave q owns hidden units u in [16q, 16q+16) for ALL 4 row-tiles and ALL 4
// gates: 16 accumulator tiles, and each weight column is read by exactly one
// wave (redundancy-1 -> 1 MB/step from L2). Cell state c stays in registers.
__global__ __launch_bounds__(512) void encoder_lstm(
    const bf16_t* __restrict__ embT,
    const bf16_t* __restrict__ WtF, const bf16_t* __restrict__ WtB,
    const float* __restrict__ bF, const float* __restrict__ bB,
    bf16_t* __restrict__ memB) {
    __shared__ bf16_t xbuf[64 * 512];           // x_t = [emb | h], bf16, 64 KB

    const int dir = blockIdx.x;
    const bf16_t* Wt  = dir ? WtB : WtF;
    const float* bias = dir ? bB : bF;
    const int tid = threadIdx.x, lane = tid & 31;
    const int q = tid >> 5;                              // wave id 0..15
    const int n0 = lane & 15, roff = (lane & 16) ? 8 : 0;
    const int u = q * 16 + n0;                           // this lane's unit

    for (int i = tid; i < 64 * 512 / 8; i += 512)
        ((uint4*)xbuf)[i] = make_uint4(0u, 0u, 0u, 0u);
    __syncthreads();   // zero-init complete before first emb copy

    float bia[4];
#pragma unroll
    for (int gate = 0; gate < 4; ++gate)
        bia[gate] = bias[gate * 256 + u];

    float creg[4][8];
#pragma unroll
    for (int m = 0; m < 4; ++m)
#pragma unroll
        for (int v = 0; v < 8; ++v) creg[m][v] = 0.0f;

    const v8f z8 = {0.f,0.f,0.f,0.f,0.f,0.f,0.f,0.f};

    for (int s = 0; s < TX_; ++s) {
        int t = dir ? (TX_ - 1 - s) : s;

        // Launder a zero *offset* (not the pointer): weight loads stay in the
        // global address space (global_load_b128, not flat_load_b128) but are
        // no longer loop-invariant, so LICM cannot hoist 1 MB into scratch.
        size_t woff = 0;
        asm volatile("" : "+s"(woff));
        const bf16_t* Wts = Wt + woff;

        // stage emb_t into xbuf[:, 0:256] with CDNA5 async global->LDS DMA
        // (h region holds previous step's h)
        {
            const bf16_t* srcb = embT + (size_t)t * B_ * DE_;
            for (int i = tid; i < B_ * DE_ / 8; i += 512) {
                int b = i >> 5, c4 = i & 31;
                unsigned dst = (unsigned)(size_t)(xbuf + b * 512) + (unsigned)(c4 * 16);
                const bf16_t* g = srcb + (size_t)i * 8;
                asm volatile("global_load_async_to_lds_b128 %0, %1, off"
                             :: "v"(dst), "v"(g) : "memory");
            }
            asm volatile("s_wait_asynccnt 0x0" ::: "memory");
        }
        __syncthreads();

        v8f acc[4][4];                       // [m][gate]
#pragma unroll
        for (int m = 0; m < 4; ++m)
#pragma unroll
            for (int gate = 0; gate < 4; ++gate) acc[m][gate] = z8;

        for (int kc = 0; kc < 16; ++kc) {
            v16bf a0 = load_frag(xbuf + (0 * 16) * 512 + kc * 32, 512);
            v16bf a1 = load_frag(xbuf + (1 * 16) * 512 + kc * 32, 512);
            v16bf a2 = load_frag(xbuf + (2 * 16) * 512 + kc * 32, 512);
            v16bf a3 = load_frag(xbuf + (3 * 16) * 512 + kc * 32, 512);
#pragma unroll
            for (int gate = 0; gate < 4; ++gate) {
                int ncol = gate * 256 + q * 16;
                v16bf bb = load_frag(Wts + (size_t)ncol * 512 + kc * 32, 512);
                acc[0][gate] = wmma_bf(a0, bb, acc[0][gate]);
                acc[1][gate] = wmma_bf(a1, bb, acc[1][gate]);
                acc[2][gate] = wmma_bf(a2, bb, acc[2][gate]);
                acc[3][gate] = wmma_bf(a3, bb, acc[3][gate]);
            }
        }
        __syncthreads();   // all waves done reading h_{t-1} before writing h_t

#pragma unroll
        for (int m = 0; m < 4; ++m)
#pragma unroll
            for (int v = 0; v < 8; ++v) {
                int b = m * 16 + v + roff;
                float iv = sigmoidf_(acc[m][0][v] + bia[0]);
                float jv = tanhf   (acc[m][1][v] + bia[1]);
                float fv = sigmoidf_(acc[m][2][v] + bia[2] + 1.0f);
                float ov = sigmoidf_(acc[m][3][v] + bia[3]);
                float cv = creg[m][v] * fv + iv * jv;
                creg[m][v] = cv;
                float hv = tanhf(cv) * ov;
                memB[((size_t)t * B_ + b) * (2 * HE_) + dir * HE_ + u] = (bf16_t)hv;
                xbuf[b * 512 + 256 + u] = (bf16_t)hv;
            }
        // next iteration's __syncthreads (after emb copy) orders h writes
    }
}

// -------------------------- keys = memory @ W_k ----------------------------
// A = memB [TX*B][512] bf16, B = WtK [256][512], C = keys f32 [TX*B][256]
__global__ __launch_bounds__(256) void keys_gemm(
    const bf16_t* __restrict__ memB, const bf16_t* __restrict__ WtK,
    float* __restrict__ keys) {
    const int tid = threadIdx.x, lane = tid & 31;
    const int Wv = blockIdx.x * 8 + (tid >> 5);   // 0..32767
    const int tm = Wv >> 4, tn = Wv & 15;
    const v8f z8 = {0.f,0.f,0.f,0.f,0.f,0.f,0.f,0.f};
    v8f acc = z8;
    const bf16_t* ap = memB + (size_t)(tm * 16) * 512;
    const bf16_t* bp = WtK + (size_t)(tn * 16) * 512;
    for (int kc = 0; kc < 16; ++kc) {
        __builtin_prefetch(ap + (kc + 2) * 32, 0, 1);
        v16bf a = load_frag(ap + kc * 32, 512);
        v16bf bb = load_frag(bp + kc * 32, 512);
        acc = wmma_bf(a, bb, acc);
    }
    const int n0 = lane & 15, roff = (lane & 16) ? 8 : 0;
#pragma unroll
    for (int v = 0; v < 8; ++v)
        keys[(size_t)(tm * 16 + v + roff) * AU_ + tn * 16 + n0] = acc[v];
}

// --------------------------- decoder init ----------------------------------
__global__ void dec_init(float* __restrict__ cbuf, bf16_t* __restrict__ x0,
                         const bf16_t* __restrict__ dembT) {
    int i = blockIdx.x * 256 + threadIdx.x;     // 65536 total
    if (i < B_ * HD_) cbuf[i] = 0.0f;
    int b = i >> 10, c = i & 1023;
    x0[i] = (c < DE_) ? dembT[(size_t)b * DE_ + c] : (bf16_t)0.0f;
}

// --------------------------- decoder LSTM cell -----------------------------
// z = x_t @ W_dec, x_t = [emb|attn|h] (K=1024), N=2048.  128 waves (16 WGs).
__global__ __launch_bounds__(256) void dec_cell(
    const bf16_t* __restrict__ xcur, bf16_t* __restrict__ xnext,
    const bf16_t* __restrict__ Wt, const float* __restrict__ bias,
    float* __restrict__ cbuf, bf16_t* __restrict__ hB) {
    const int tid = threadIdx.x, lane = tid & 31;
    const int Wv = blockIdx.x * 8 + (tid >> 5);  // 0..127
    const int m = Wv & 3, g = Wv >> 2;           // g in 0..31
    const int n0 = lane & 15, roff = (lane & 16) ? 8 : 0;
    const v8f z8 = {0.f,0.f,0.f,0.f,0.f,0.f,0.f,0.f};
    v8f acc[4] = {z8, z8, z8, z8};
    for (int kc = 0; kc < 32; ++kc) {
        v16bf a = load_frag(xcur + (size_t)(m * 16) * 1024 + kc * 32, 1024);
#pragma unroll
        for (int gate = 0; gate < 4; ++gate) {
            int ncol = gate * HD_ + g * 16;
            v16bf bb = load_frag(Wt + (size_t)ncol * 1024 + kc * 32, 1024);
            acc[gate] = wmma_bf(a, bb, acc[gate]);
        }
    }
    const int u = g * 16 + n0;
    float bi = bias[0 * HD_ + u], bj = bias[1 * HD_ + u];
    float bf = bias[2 * HD_ + u], bo = bias[3 * HD_ + u];
#pragma unroll
    for (int v = 0; v < 8; ++v) {
        int b = m * 16 + v + roff;
        float iv = sigmoidf_(acc[0][v] + bi);
        float jv = tanhf   (acc[1][v] + bj);
        float fv = sigmoidf_(acc[2][v] + bf + 1.0f);
        float ov = sigmoidf_(acc[3][v] + bo);
        float cv = cbuf[b * HD_ + u] * fv + iv * jv;
        cbuf[b * HD_ + u] = cv;
        float hv = tanhf(cv) * ov;
        hB[b * HD_ + u] = (bf16_t)hv;
        xnext[b * 1024 + 512 + u] = (bf16_t)hv;
    }
}

// ------------------------------ pq = h @ W_q -------------------------------
__global__ __launch_bounds__(256) void pq_gemm(
    const bf16_t* __restrict__ hB, const bf16_t* __restrict__ WtQ,
    float* __restrict__ pq) {
    const int tid = threadIdx.x, lane = tid & 31;
    const int Wv = blockIdx.x * 8 + (tid >> 5);  // 0..63
    const int m = Wv & 3, tn = Wv >> 2;
    const v8f z8 = {0.f,0.f,0.f,0.f,0.f,0.f,0.f,0.f};
    v8f acc = z8;
    for (int kc = 0; kc < 16; ++kc) {
        v16bf a = load_frag(hB + (size_t)(m * 16) * HD_ + kc * 32, HD_);
        v16bf bb = load_frag(WtQ + (size_t)(tn * 16) * HD_ + kc * 32, HD_);
        acc = wmma_bf(a, bb, acc);
    }
    const int n0 = lane & 15, roff = (lane & 16) ? 8 : 0;
#pragma unroll
    for (int v = 0; v < 8; ++v)
        pq[(m * 16 + v + roff) * AU_ + tn * 16 + n0] = acc[v];
}

// --------------- score -> softmax -> context -> abuf = [h|ctx] -------------
// one block per batch row b
__global__ __launch_bounds__(256) void score_ctx(
    const float* __restrict__ keys, const float* __restrict__ pq,
    const float* __restrict__ vatt, const bf16_t* __restrict__ memB,
    const bf16_t* __restrict__ hB, bf16_t* __restrict__ abuf) {
    const int b = blockIdx.x, tid = threadIdx.x;
    __shared__ float sc[TX_];
    __shared__ float red[256];
    __shared__ float pqs[AU_];
    __shared__ float vs[AU_];
    pqs[tid] = pq[b * AU_ + tid];
    vs[tid]  = vatt[tid];
    __syncthreads();
    for (int s = tid; s < TX_; s += 256) {
        const float* kr = keys + ((size_t)s * B_ + b) * AU_;
        float a = 0.0f;
        for (int u = 0; u < AU_; ++u) a += tanhf(kr[u] + pqs[u]) * vs[u];
        sc[s] = a;
    }
    __syncthreads();
    red[tid] = fmaxf(sc[tid], sc[tid + 256]);
    __syncthreads();
    for (int st = 128; st > 0; st >>= 1) {
        if (tid < st) red[tid] = fmaxf(red[tid], red[tid + st]);
        __syncthreads();
    }
    float mx = red[0];
    __syncthreads();
    float e0 = __expf(sc[tid] - mx), e1 = __expf(sc[tid + 256] - mx);
    sc[tid] = e0; sc[tid + 256] = e1;
    red[tid] = e0 + e1;
    __syncthreads();
    for (int st = 128; st > 0; st >>= 1) {
        if (tid < st) red[tid] += red[tid + st];
        __syncthreads();
    }
    float inv = 1.0f / red[0];
    __syncthreads();
#pragma unroll
    for (int d0 = 0; d0 < 2; ++d0) {
        int d = tid + 256 * d0;
        float cx = 0.0f;
        for (int s = 0; s < TX_; ++s)
            cx += sc[s] * (float)memB[((size_t)s * B_ + b) * (2 * HE_) + d];
        abuf[b * 1024 + 512 + d] = (bf16_t)(cx * inv);
    }
    abuf[b * 1024 + tid]       = hB[b * HD_ + tid];
    abuf[b * 1024 + tid + 256] = hB[b * HD_ + tid + 256];
}

// --------------- attn = [h|ctx] @ W_attn, feed next x, save for logits -----
__global__ __launch_bounds__(256) void attn_gemm(
    const bf16_t* __restrict__ abuf, const bf16_t* __restrict__ WtA,
    bf16_t* __restrict__ attnB, bf16_t* __restrict__ xnext,
    const bf16_t* __restrict__ dembT, int t) {
    const int tid = threadIdx.x, lane = tid & 31;
    const int Wv = blockIdx.x * 8 + (tid >> 5);  // 0..63
    const int m = Wv & 3, tn = Wv >> 2;
    const v8f z8 = {0.f,0.f,0.f,0.f,0.f,0.f,0.f,0.f};
    v8f acc = z8;
    for (int kc = 0; kc < 32; ++kc) {
        v16bf a = load_frag(abuf + (size_t)(m * 16) * 1024 + kc * 32, 1024);
        v16bf bb = load_frag(WtA + (size_t)(tn * 16) * 1024 + kc * 32, 1024);
        acc = wmma_bf(a, bb, acc);
    }
    const int n0 = lane & 15, roff = (lane & 16) ? 8 : 0;
#pragma unroll
    for (int v = 0; v < 8; ++v) {
        int b = m * 16 + v + roff;
        int n = tn * 16 + n0;
        bf16_t hv = (bf16_t)acc[v];
        attnB[((size_t)t * B_ + b) * AT_ + n] = hv;
        if (t + 1 < TY_) xnext[b * 1024 + 256 + n] = hv;
    }
    if (t + 1 < TY_) {   // stage next step's embedding into xnext[:, 0:256]
        int gi = blockIdx.x * 256 + tid;  // 0..2047
        int b = gi >> 5, c4 = gi & 31;
        const uint4* src = (const uint4*)(dembT + (size_t)(t + 1) * B_ * DE_);
        ((uint4*)(xnext + (size_t)b * 1024))[c4] = src[gi];
    }
}

// ----------------- logits = attn @ W_proj + b, [B,TY,V] --------------------
__global__ __launch_bounds__(256) void logits_gemm(
    const bf16_t* __restrict__ attnB, const bf16_t* __restrict__ WtP,
    const float* __restrict__ bproj, float* __restrict__ out) {
    const int tid = threadIdx.x, lane = tid & 31;
    const long long Wv = (long long)blockIdx.x * 8 + (tid >> 5);
    const int tm = (int)(Wv / NTP), tn = (int)(Wv % NTP);
    const v8f z8 = {0.f,0.f,0.f,0.f,0.f,0.f,0.f,0.f};
    v8f acc = z8;
    const bf16_t* ap = attnB + (size_t)(tm * 16) * AT_;
    const bf16_t* bp = WtP + (size_t)(tn * 16) * AT_;
    for (int kc = 0; kc < 8; ++kc) {
        __builtin_prefetch(bp + (kc + 2) * 32, 0, 1);
        v16bf a = load_frag(ap + kc * 32, AT_);
        v16bf bb = load_frag(bp + kc * 32, AT_);
        acc = wmma_bf(a, bb, acc);
    }
    const int n0 = lane & 15, roff = (lane & 16) ? 8 : 0;
#pragma unroll
    for (int v = 0; v < 8; ++v) {
        int r = tm * 16 + v + roff;      // r = t*64 + b
        int t = r >> 6, b = r & 63;
        int n = tn * 16 + n0;
        if (n < V_)
            out[((size_t)b * TY_ + t) * V_ + n] = acc[v] + bproj[n];
    }
}

// ------------------------------ host side ----------------------------------
extern "C" void kernel_launch(void* const* d_in, const int* in_sizes, int n_in,
                              void* d_out, int out_size, void* d_ws, size_t ws_size,
                              hipStream_t stream) {
    const int*   enc_ids = (const int*)d_in[0];
    const int*   dec_ids = (const int*)d_in[1];
    const float* E       = (const float*)d_in[2];
    const float* W_fwd   = (const float*)d_in[3];
    const float* b_fwd   = (const float*)d_in[4];
    const float* W_bwd   = (const float*)d_in[5];
    const float* b_bwd   = (const float*)d_in[6];
    const float* W_dec   = (const float*)d_in[7];
    const float* b_dec   = (const float*)d_in[8];
    const float* W_k     = (const float*)d_in[9];
    const float* W_q     = (const float*)d_in[10];
    const float* v_att   = (const float*)d_in[11];
    const float* W_attn  = (const float*)d_in[12];
    const float* W_proj  = (const float*)d_in[13];
    const float* b_proj  = (const float*)d_in[14];
    float* out = (float*)d_out;
    (void)in_sizes; (void)n_in; (void)out_size; (void)ws_size;

    char* p = (char*)d_ws;
    auto alloc = [&](size_t elems, size_t esz) -> char* {
        char* r = p;
        p += (elems * esz + 255) & ~(size_t)255;
        return r;
    };
    // ~109 MB of workspace total
    bf16_t* WtF   = (bf16_t*)alloc((size_t)1024 * 512, 2);
    bf16_t* WtB   = (bf16_t*)alloc((size_t)1024 * 512, 2);
    bf16_t* WtD   = (bf16_t*)alloc((size_t)2048 * 1024, 2);
    bf16_t* WtK   = (bf16_t*)alloc((size_t)256 * 512, 2);
    bf16_t* WtQ   = (bf16_t*)alloc((size_t)256 * 512, 2);
    bf16_t* WtA   = (bf16_t*)alloc((size_t)256 * 1024, 2);
    bf16_t* WtP   = (bf16_t*)alloc((size_t)VPAD * 256, 2);
    bf16_t* embE  = (bf16_t*)alloc((size_t)TX_ * B_ * DE_, 2);
    bf16_t* embD  = (bf16_t*)alloc((size_t)TY_ * B_ * DE_, 2);
    bf16_t* memB  = (bf16_t*)alloc((size_t)TX_ * B_ * 2 * HE_, 2);
    float*  keys  = (float*) alloc((size_t)TX_ * B_ * AU_, 4);
    float*  cbuf  = (float*) alloc((size_t)B_ * HD_, 4);
    bf16_t* hB    = (bf16_t*)alloc((size_t)B_ * HD_, 2);
    bf16_t* xb0   = (bf16_t*)alloc((size_t)B_ * 1024, 2);
    bf16_t* xb1   = (bf16_t*)alloc((size_t)B_ * 1024, 2);
    float*  pqb   = (float*) alloc((size_t)B_ * AU_, 4);
    bf16_t* abuf  = (bf16_t*)alloc((size_t)B_ * 1024, 2);
    bf16_t* attnB = (bf16_t*)alloc((size_t)TY_ * B_ * AT_, 2);

    // weight prepack (f32 [K][N] -> bf16 [Npad][K])
    prepack_w<<<2048,  256, 0, stream>>>(W_fwd,  WtF, 512,  1024,  1024);
    prepack_w<<<2048,  256, 0, stream>>>(W_bwd,  WtB, 512,  1024,  1024);
    prepack_w<<<8192,  256, 0, stream>>>(W_dec,  WtD, 1024, 2048,  2048);
    prepack_w<<<512,   256, 0, stream>>>(W_k,    WtK, 512,  256,   256);
    prepack_w<<<512,   256, 0, stream>>>(W_q,    WtQ, 512,  256,   256);
    prepack_w<<<1024,  256, 0, stream>>>(W_attn, WtA, 1024, 256,   256);
    prepack_w<<<10016, 256, 0, stream>>>(W_proj, WtP, 256,  10000, VPAD);

    // embeddings (time-major bf16)
    embed_g<<<32768, 256, 0, stream>>>(enc_ids, E, embE, TX_);
    embed_g<<<16384, 256, 0, stream>>>(dec_ids, E, embD, TY_);

    // persistent bidirectional encoder (1 WG per direction, 16 waves)
    encoder_lstm<<<2, 512, 0, stream>>>(embE, WtF, WtB, b_fwd, b_bwd, memB);

    // keys = memory @ W_k  (GPU-wide WMMA GEMM)
    keys_gemm<<<4096, 256, 0, stream>>>(memB, WtK, keys);

    // decoder recurrence
    dec_init<<<256, 256, 0, stream>>>(cbuf, xb0, embD);
    for (int t = 0; t < TY_; ++t) {
        bf16_t* xc = (t & 1) ? xb1 : xb0;
        bf16_t* xn = (t & 1) ? xb0 : xb1;
        dec_cell <<<16, 256, 0, stream>>>(xc, xn, WtD, b_dec, cbuf, hB);
        pq_gemm  <<<8,  256, 0, stream>>>(hB, WtQ, pqb);
        score_ctx<<<64, 256, 0, stream>>>(keys, pqb, v_att, memB, hB, abuf);
        attn_gemm<<<8,  256, 0, stream>>>(abuf, WtA, attnB, xn, embD, t);
    }

    // logits = attn @ W_proj + b_proj  ->  out[B][TY][V]
    logits_gemm<<<80128, 256, 0, stream>>>(attnB, WtP, b_proj, out);
}
